// SelectiveSSM_30030411333713
// MI455X (gfx1250) — matrix operand order repaired and checked by reading, not compiled, not run on hardware
//
#include <hip/hip_runtime.h>
#include <hip/hip_bf16.h>

// ---------------------------------------------------------------------------
// Feature detection (device pass only; host pass falls back cleanly)
// ---------------------------------------------------------------------------
#if defined(__AMDGCN__) && __has_builtin(__builtin_amdgcn_tensor_load_to_lds) && \
    __has_builtin(__builtin_amdgcn_s_wait_tensorcnt)
#define HAVE_TDM 1
#else
#define HAVE_TDM 0
#endif

#if __has_include(<hip/amd_detail/amd_gfx1250_TDM.h>)
#define TDM_SIX_ARGS 1
#else
#define TDM_SIX_ARGS 0
#endif

// ---------------------------------------------------------------------------
// Types for CDNA5 WMMA / TDM
// ---------------------------------------------------------------------------
typedef __attribute__((ext_vector_type(16))) __bf16 v16bf;
typedef __attribute__((ext_vector_type(8)))  float  v8f;
typedef __attribute__((ext_vector_type(4)))  unsigned int u32x4;
typedef __attribute__((ext_vector_type(8)))  int i32x8;
typedef __attribute__((ext_vector_type(4)))  int i32x4;

union FragU {
    uint4  u[2];
    v16bf  v;
};

// ---------------------------------------------------------------------------
// Dimensions (compile-time constants from the reference)
// ---------------------------------------------------------------------------
#define D_MODEL 1024
#define D_STATE 16
#define D_CONV  4
#define D_INNER 2048
#define BATCH   4
#define SEQ     2048
#define NROWS   (BATCH * SEQ)          // 8192
#define XPROJ_N (2 * D_STATE + 1)      // 33

// ---------------------------------------------------------------------------
// bf16 helpers (manual, round-to-nearest-even)
// ---------------------------------------------------------------------------
__device__ __forceinline__ unsigned short f2bf(float f) {
    unsigned int u = __float_as_uint(f);
    unsigned int r = (u + 0x7FFFu + ((u >> 16) & 1u)) >> 16;
    return (unsigned short)r;
}
__device__ __forceinline__ float bf2f(unsigned short h) {
    return __uint_as_float(((unsigned int)h) << 16);
}

__device__ __forceinline__ void store_out(float* C, size_t i, float v) { C[i] = v; }
__device__ __forceinline__ void store_out(unsigned short* C, size_t i, float v) { C[i] = f2bf(v); }

#if HAVE_TDM
// low 32 bits of a flat pointer to LDS == LDS byte address (ISA aperture rule)
__device__ __forceinline__ unsigned lds_addr_of(const void* p) {
    return (unsigned)(unsigned long long)(uintptr_t)p;
}

// Issue a TDM 2D tile load: tile (tileX elems contiguous) x (tileY rows),
// bf16 elements (data_size = 2B), global row stride = strideElems.
// D# layout per CDNA5 ISA ch. 8 (groups 2/3 zero => 2D tensor).
__device__ __forceinline__ void tdm_load_tile_2d(const void* gptr, unsigned ldsAddr,
                                                 int tileX, int tileY,
                                                 long long strideElems) {
    unsigned long long ga = (unsigned long long)(uintptr_t)gptr;
    u32x4 g0;
    g0[0] = 1u;                                          // count=1 (valid descriptor)
    g0[1] = ldsAddr;                                     // lds_addr (bytes)
    g0[2] = (unsigned)(ga & 0xFFFFFFFFull);              // global_addr[31:0]
    g0[3] = (unsigned)((ga >> 32) & 0x01FFFFFFull)       // global_addr[56:32]
            | (2u << 30);                                // type = 2 ("image")
    i32x8 g1;
    g1[0] = (int)(1u << 16);                             // data_size=1 -> 2 bytes
    g1[1] = (int)(((unsigned)tileX & 0xFFFFu) << 16);    // tensor_dim0[15:0]
    g1[2] = (int)((((unsigned)tileX >> 16) & 0xFFFFu)    // tensor_dim0[31:16]
            | (((unsigned)tileY & 0xFFFFu) << 16));      // tensor_dim1[15:0]
    g1[3] = (int)((((unsigned)tileY >> 16) & 0xFFFFu)    // tensor_dim1[31:16]
            | (((unsigned)tileX & 0xFFFFu) << 16));      // tile_dim0
    g1[4] = (int)((unsigned)tileY & 0xFFFFu);            // tile_dim1 (tile_dim2=0)
    g1[5] = (int)(strideElems & 0xFFFFFFFFll);           // tensor_dim0_stride[31:0]
    g1[6] = (int)((strideElems >> 32) & 0xFFFFll);       // tensor_dim0_stride[47:32]
    g1[7] = 0;
    i32x4 z4 = {0, 0, 0, 0};
#if TDM_SIX_ARGS
    i32x8 z8 = {0, 0, 0, 0, 0, 0, 0, 0};
    __builtin_amdgcn_tensor_load_to_lds(g0, g1, z4, z4, z8, 0);
#else
    __builtin_amdgcn_tensor_load_to_lds(g0, g1, z4, z4, 0);
#endif
}
#endif  // HAVE_TDM

// ---------------------------------------------------------------------------
// fp32 -> bf16 conversion (plain)
// ---------------------------------------------------------------------------
__global__ void f32_to_bf16_kernel(const float* __restrict__ in,
                                   unsigned short* __restrict__ out, int n) {
    int i = blockIdx.x * blockDim.x + threadIdx.x;
    if (i < n) out[i] = f2bf(in[i]);
}

// ---------------------------------------------------------------------------
// fp32 [K][N] -> bf16 transposed [N][K]  (one-time weight prep; LDS tiled)
// Grid: (N/32, K/32), block (32, 8).
// ---------------------------------------------------------------------------
__global__ void transpose_f32_to_bf16(const float* __restrict__ in,       // [K][N]
                                      unsigned short* __restrict__ out,   // [N][K]
                                      int K, int N) {
    __shared__ unsigned short tile[32][33];
    const int n0 = blockIdx.x * 32;
    const int k0 = blockIdx.y * 32;
    const int tx = threadIdx.x;   // 0..31
    const int ty = threadIdx.y;   // 0..7
#pragma unroll
    for (int i = 0; i < 4; ++i) {
        int k = k0 + ty + i * 8;
        tile[ty + i * 8][tx] = f2bf(in[(size_t)k * N + n0 + tx]);
    }
    __syncthreads();
#pragma unroll
    for (int i = 0; i < 4; ++i) {
        int n = n0 + ty + i * 8;
        out[(size_t)n * K + k0 + tx] = tile[tx][ty + i * 8];
    }
}

// ---------------------------------------------------------------------------
// Tiled bf16 WMMA GEMM: C[M,N] = A[M,K] * BT[N,K]^T  (both operands K-major).
// Block = 256 threads (8 waves). Block tile 128(M) x 128(N), K-step 32.
// Wave (mg = w&3, ng = w>>2) computes M rows [mg*32,+32) x N cols [ng*64,+64)
// = 2x4 WMMA tiles. Double-buffered LDS; tiles arrive either via the Tensor
// Data Mover (one wave issues tensor_load_to_lds, TENSORcnt-synced) or via
// register-staged global loads (fallback when the TDM builtin is absent).
// ---------------------------------------------------------------------------
template <typename OutT>
__global__ void gemm_bf16_wmma(const unsigned short* __restrict__ A,    // [M][K]
                               const unsigned short* __restrict__ BT,   // [N][K]
                               OutT* __restrict__ C,
                               int M, int N, int K) {
    __shared__ uint4 sA[2][128 * 4];   // [buf][row][k-chunk] : 128x32 bf16
    __shared__ uint4 sB[2][128 * 4];   // [buf][n][k-chunk]   : 128x32 bf16

    const int tid  = threadIdx.x;       // 0..255
    const int wave = tid >> 5;          // 0..7
    const int lane = tid & 31;
    const int half = (lane >> 4) & 1;   // lane 0-15 -> 0, 16-31 -> 1
    const int r    = lane & 15;
    const int mg   = wave & 3;          // M sub-block (32 rows)
    const int ng   = wave >> 2;         // N sub-block (64 cols)

    const int blockM = blockIdx.y * 128;
    const int blockN = blockIdx.x * 128;

    v8f acc[2][4] = {};
    int cur = 0;

#if HAVE_TDM
    // ---------------- TDM-staged pipeline ----------------
    if (wave == 0) {
        tdm_load_tile_2d(A  + (size_t)blockM * K, lds_addr_of(&sA[0][0]), 32, 128, K);
        tdm_load_tile_2d(BT + (size_t)blockN * K, lds_addr_of(&sB[0][0]), 32, 128, K);
    }
    for (int k0 = 0; k0 < K; k0 += 32) {
        if (wave == 0) __builtin_amdgcn_s_wait_tensorcnt(0);  // buf[cur] complete
        __syncthreads();                                      // ...and visible to all
        if (k0 + 32 < K && wave == 0) {
            tdm_load_tile_2d(A  + (size_t)blockM * K + (k0 + 32),
                             lds_addr_of(&sA[cur ^ 1][0]), 32, 128, K);
            tdm_load_tile_2d(BT + (size_t)blockN * K + (k0 + 32),
                             lds_addr_of(&sB[cur ^ 1][0]), 32, 128, K);
        }
        FragU a[2], b[4];
#pragma unroll
        for (int s = 0; s < 2; ++s) {
            int row = mg * 32 + s * 16 + r;
            a[s].u[0] = sA[cur][row * 4 + half];
            a[s].u[1] = sA[cur][row * 4 + 2 + half];
        }
#pragma unroll
        for (int t = 0; t < 4; ++t) {
            int n = ng * 64 + t * 16 + r;
            b[t].u[0] = sB[cur][n * 4 + half];
            b[t].u[1] = sB[cur][n * 4 + 2 + half];
        }
#pragma unroll
        for (int s = 0; s < 2; ++s)
#pragma unroll
            for (int t = 0; t < 4; ++t)
                acc[s][t] = __builtin_amdgcn_wmma_f32_16x16x32_bf16(
                    false, a[s].v, false, b[t].v, (short)0, acc[s][t], false, false);
        cur ^= 1;
    }
#else
    // ---------------- register-staged fallback ----------------
    uint4 ra[2], rb[2];
#pragma unroll
    for (int i = 0; i < 2; ++i) {
        int u = tid + i * 256;
        int row = u >> 2;
        int ch  = u & 3;
        ra[i] = *reinterpret_cast<const uint4*>(A  + (size_t)(blockM + row) * K + ch * 8);
        rb[i] = *reinterpret_cast<const uint4*>(BT + (size_t)(blockN + row) * K + ch * 8);
    }
    for (int k0 = 0; k0 < K; k0 += 32) {
#pragma unroll
        for (int i = 0; i < 2; ++i) {
            int u = tid + i * 256;
            sA[cur][u] = ra[i];
            sB[cur][u] = rb[i];
        }
        __syncthreads();
        if (k0 + 32 < K) {
#pragma unroll
            for (int i = 0; i < 2; ++i) {
                int u = tid + i * 256;
                int row = u >> 2;
                int ch  = u & 3;
                ra[i] = *reinterpret_cast<const uint4*>(
                    A + (size_t)(blockM + row) * K + (k0 + 32) + ch * 8);
                rb[i] = *reinterpret_cast<const uint4*>(
                    BT + (size_t)(blockN + row) * K + (k0 + 32) + ch * 8);
            }
        }
        if (k0 + 64 < K) {
            int row = tid >> 1;
            __builtin_prefetch(A  + (size_t)(blockM + row) * K + (k0 + 64) + (tid & 1) * 16, 0, 0);
            __builtin_prefetch(BT + (size_t)(blockN + row) * K + (k0 + 64) + (tid & 1) * 16, 0, 0);
        }
        FragU a[2], b[4];
#pragma unroll
        for (int s = 0; s < 2; ++s) {
            int row = mg * 32 + s * 16 + r;
            a[s].u[0] = sA[cur][row * 4 + half];
            a[s].u[1] = sA[cur][row * 4 + 2 + half];
        }
#pragma unroll
        for (int t = 0; t < 4; ++t) {
            int n = ng * 64 + t * 16 + r;
            b[t].u[0] = sB[cur][n * 4 + half];
            b[t].u[1] = sB[cur][n * 4 + 2 + half];
        }
#pragma unroll
        for (int s = 0; s < 2; ++s)
#pragma unroll
            for (int t = 0; t < 4; ++t)
                acc[s][t] = __builtin_amdgcn_wmma_f32_16x16x32_bf16(
                    false, a[s].v, false, b[t].v, (short)0, acc[s][t], false, false);
        cur ^= 1;
    }
#endif

    // C/D layout: VGPR rr -> M = rr + half*8 ; N = lane%16
#pragma unroll
    for (int s = 0; s < 2; ++s) {
#pragma unroll
        for (int t = 0; t < 4; ++t) {
#pragma unroll
            for (int rr = 0; rr < 8; ++rr) {
                int row = blockM + mg * 32 + s * 16 + rr + half * 8;
                int col = blockN + ng * 64 + t * 16 + r;
                store_out(C, (size_t)row * N + col, acc[s][t][rr]);
            }
        }
    }
}

// ---------------------------------------------------------------------------
// Depthwise causal conv (k=4) + bias + SiLU. x_proj = xz[:, 0:2048].
// ---------------------------------------------------------------------------
__global__ void conv_silu_kernel(const unsigned short* __restrict__ xz,  // [8192, 4096]
                                 const float* __restrict__ cw,           // [2048, 1, 4]
                                 const float* __restrict__ cb,           // [2048]
                                 unsigned short* __restrict__ xc) {      // [8192, 2048]
    int idx = blockIdx.x * blockDim.x + threadIdx.x;
    if (idx >= BATCH * SEQ * D_INNER) return;
    int c = idx & (D_INNER - 1);
    int l = (idx >> 11) & (SEQ - 1);
    int b = idx >> 22;
    float acc = cb[c];
    size_t base = (size_t)b * SEQ * (2 * D_INNER);
#pragma unroll
    for (int t = 0; t < D_CONV; ++t) {
        int ll = l - (D_CONV - 1) + t;
        if (ll >= 0)
            acc += cw[c * D_CONV + t] * bf2f(xz[base + (size_t)ll * (2 * D_INNER) + c]);
    }
    float s = acc / (1.0f + __expf(-acc));   // silu
    xc[(size_t)idx] = f2bf(s);
}

// ---------------------------------------------------------------------------
// ssm = x_conv @ W_xproj  (N = 33, tiny; scalar, L2-resident operands)
// ---------------------------------------------------------------------------
__global__ void xproj_kernel(const unsigned short* __restrict__ xc,  // [8192, 2048]
                             const float* __restrict__ W,            // [2048, 33]
                             float* __restrict__ ssm) {              // [8192, 33]
    int idx = blockIdx.x * blockDim.x + threadIdx.x;
    if (idx >= NROWS * XPROJ_N) return;
    int row = idx / XPROJ_N;
    int j   = idx - row * XPROJ_N;
    const unsigned short* xr = xc + (size_t)row * D_INNER;
    float acc = 0.0f;
#pragma unroll 8
    for (int k = 0; k < D_INNER; ++k)
        acc += bf2f(xr[k]) * W[k * XPROJ_N + j];
    ssm[idx] = acc;
}

// ---------------------------------------------------------------------------
// delta = softplus(ssm[...,0]); deltaB = delta * B; C extraction. (bf16 out)
// ---------------------------------------------------------------------------
__global__ void gate_prep_kernel(const float* __restrict__ ssm,       // [8192, 33]
                                 unsigned short* __restrict__ dB,     // [8192, 16]
                                 unsigned short* __restrict__ Cm) {   // [8192, 16]
    int idx = blockIdx.x * blockDim.x + threadIdx.x;
    if (idx >= NROWS * D_STATE) return;
    int row = idx >> 4;
    int n   = idx & 15;
    const float* s = ssm + (size_t)row * XPROJ_N;
    float d0 = s[0];
    float delta = (d0 > 20.0f) ? d0 : log1pf(__expf(d0));  // softplus
    dB[idx] = f2bf(delta * s[1 + n]);
    Cm[idx] = f2bf(s[1 + D_STATE + n]);
}

// ---------------------------------------------------------------------------
// h_mean[b,d,n] = (1/L) * sum_l x_conv[b,l,d] * deltaB[b,l,n]
// WMMA GEMM with transposed A staging. M=2048(d), N=16(n), K=2048(l) per batch.
// Block = 128 threads (4 waves), handles 64 d rows; grid = (32, BATCH).
// ---------------------------------------------------------------------------
__global__ void hmean_wmma_kernel(const unsigned short* __restrict__ xc,  // [8192, 2048]
                                  const unsigned short* __restrict__ dB,  // [8192, 16]
                                  float* __restrict__ hm) {               // [B, 2048, 16]
    __shared__ __align__(16) unsigned short sXt[64 * 32];  // [d][l]
    __shared__ __align__(16) unsigned short sBt[16 * 32];  // [n][l]

    const int tid  = threadIdx.x;
    const int wave = tid >> 5;
    const int lane = tid & 31;
    const int half = (lane >> 4) & 1;
    const int r    = lane & 15;

    const int mb    = blockIdx.x;   // 0..31
    const int b     = blockIdx.y;   // 0..3
    const int dBase = mb * 64;

    v8f acc = {};

    for (int l0 = 0; l0 < SEQ; l0 += 32) {
#pragma unroll
        for (int i = 0; i < 8; ++i) {
            int u  = tid + i * 128;
            int ll = u >> 5;
            int dp = u & 31;
            const unsigned int* gp = reinterpret_cast<const unsigned int*>(
                xc + ((size_t)(b * SEQ + l0 + ll)) * D_INNER + dBase + dp * 2);
            unsigned int v = *gp;
            sXt[(dp * 2 + 0) * 32 + ll] = (unsigned short)(v & 0xFFFFu);
            sXt[(dp * 2 + 1) * 32 + ll] = (unsigned short)(v >> 16);
        }
#pragma unroll
        for (int i = 0; i < 2; ++i) {
            int u  = tid + i * 128;
            int ll = u >> 3;
            int np = u & 7;
            const unsigned int* gp = reinterpret_cast<const unsigned int*>(
                dB + ((size_t)(b * SEQ + l0 + ll)) * D_STATE + np * 2);
            unsigned int v = *gp;
            sBt[(np * 2 + 0) * 32 + ll] = (unsigned short)(v & 0xFFFFu);
            sBt[(np * 2 + 1) * 32 + ll] = (unsigned short)(v >> 16);
        }
        __syncthreads();

        const uint4* sXv = reinterpret_cast<const uint4*>(sXt);
        const uint4* sBv = reinterpret_cast<const uint4*>(sBt);
        FragU a, bb;
        int arow = wave * 16 + r;
        a.u[0]  = sXv[arow * 4 + half];
        a.u[1]  = sXv[arow * 4 + 2 + half];
        bb.u[0] = sBv[r * 4 + half];
        bb.u[1] = sBv[r * 4 + 2 + half];
        acc = __builtin_amdgcn_wmma_f32_16x16x32_bf16(
            false, a.v, false, bb.v, (short)0, acc, false, false);
        __syncthreads();
    }

    const float invL = 1.0f / (float)SEQ;
#pragma unroll
    for (int rr = 0; rr < 8; ++rr) {
        int d = dBase + wave * 16 + rr + half * 8;
        hm[((size_t)b * D_INNER + d) * D_STATE + r] = acc[rr] * invL;
    }
}

// ---------------------------------------------------------------------------
// y[b,l,d] = (sum_n h_mean[b,d,n]*Cm[b,l,n] + D[d]*x_conv[b,l,d]) * silu(z)
// ---------------------------------------------------------------------------
__global__ void y_gate_kernel(const float* __restrict__ hm,            // [B, 2048, 16]
                              const unsigned short* __restrict__ Cm,   // [8192, 16]
                              const unsigned short* __restrict__ xc,   // [8192, 2048]
                              const float* __restrict__ Dv,            // [2048]
                              const unsigned short* __restrict__ xz,   // [8192, 4096]
                              unsigned short* __restrict__ y) {        // [8192, 2048]
    int idx = blockIdx.x * blockDim.x + threadIdx.x;
    if (idx >= BATCH * SEQ * D_INNER) return;
    int d = idx & (D_INNER - 1);
    int l = (idx >> 11) & (SEQ - 1);
    int b = idx >> 22;
    int row = b * SEQ + l;
    const float* h = hm + ((size_t)b * D_INNER + d) * D_STATE;
    const unsigned short* cm = Cm + (size_t)row * D_STATE;
    float acc = 0.0f;
#pragma unroll
    for (int n = 0; n < D_STATE; ++n)
        acc += h[n] * bf2f(cm[n]);
    acc += Dv[d] * bf2f(xc[(size_t)idx]);
    float z = bf2f(xz[(size_t)row * (2 * D_INNER) + D_INNER + d]);
    acc *= z / (1.0f + __expf(-z));   // * silu(z)
    y[(size_t)idx] = f2bf(acc);
}

// ---------------------------------------------------------------------------
// Launcher
// ---------------------------------------------------------------------------
extern "C" void kernel_launch(void* const* d_in, const int* in_sizes, int n_in,
                              void* d_out, int out_size, void* d_ws, size_t ws_size,
                              hipStream_t stream) {
    const float* x      = (const float*)d_in[0];  // [4,2048,1024]
    const float* W_in   = (const float*)d_in[1];  // [1024,4096]
    const float* conv_w = (const float*)d_in[2];  // [2048,1,4]
    const float* conv_b = (const float*)d_in[3];  // [2048]
    const float* W_xp   = (const float*)d_in[4];  // [2048,33]
    // d_in[5] = A_log (unused by the reference computation)
    const float* Dv     = (const float*)d_in[6];  // [2048]
    const float* W_out  = (const float*)d_in[7];  // [2048,1024]
    float* out = (float*)d_out;                   // [4,2048,1024]

    // workspace layout (256-byte aligned slices)
    char* ws = (char*)d_ws;
    size_t off = 0;
    auto take = [&](size_t bytes) -> char* {
        char* p = ws + off;
        off += (bytes + 255) & ~(size_t)255;
        return p;
    };
    unsigned short* xBf    = (unsigned short*)take((size_t)NROWS * D_MODEL * 2);        // 16 MB
    unsigned short* WinT   = (unsigned short*)take((size_t)(2 * D_INNER) * D_MODEL * 2);// 8 MB  [4096][1024]
    unsigned short* WoutT  = (unsigned short*)take((size_t)D_MODEL * D_INNER * 2);      // 4 MB  [1024][2048]
    unsigned short* xzBf   = (unsigned short*)take((size_t)NROWS * 2 * D_INNER * 2);    // 64 MB
    unsigned short* xcBf   = (unsigned short*)take((size_t)NROWS * D_INNER * 2);        // 32 MB
    float*          ssmF   = (float*)take((size_t)NROWS * XPROJ_N * 4);                 // ~1 MB
    unsigned short* dBBf   = (unsigned short*)take((size_t)NROWS * D_STATE * 2);
    unsigned short* CmBf   = (unsigned short*)take((size_t)NROWS * D_STATE * 2);
    float*          hmF    = (float*)take((size_t)BATCH * D_INNER * D_STATE * 4);
    unsigned short* yBf    = (unsigned short*)take((size_t)NROWS * D_INNER * 2);        // 32 MB

    const int TPB = 256;

    // 1) conversions: x plain; W_in / W_out converted + transposed (K-major)
    {
        int n = NROWS * D_MODEL;
        f32_to_bf16_kernel<<<(n + TPB - 1) / TPB, TPB, 0, stream>>>(x, xBf, n);
    }
    transpose_f32_to_bf16<<<dim3((2 * D_INNER) / 32, D_MODEL / 32), dim3(32, 8), 0, stream>>>(
        W_in, WinT, D_MODEL, 2 * D_INNER);
    transpose_f32_to_bf16<<<dim3(D_MODEL / 32, D_INNER / 32), dim3(32, 8), 0, stream>>>(
        W_out, WoutT, D_INNER, D_MODEL);

    // 2) xz = x @ W_in  (M=8192, N=4096, K=1024), bf16 out
    gemm_bf16_wmma<unsigned short>
        <<<dim3((2 * D_INNER) / 128, NROWS / 128), 256, 0, stream>>>(
            xBf, WinT, xzBf, NROWS, 2 * D_INNER, D_MODEL);

    // 3) depthwise causal conv + bias + SiLU
    {
        int n = BATCH * SEQ * D_INNER;
        conv_silu_kernel<<<(n + TPB - 1) / TPB, TPB, 0, stream>>>(xzBf, conv_w, conv_b, xcBf);
    }

    // 4) ssm = x_conv @ W_xproj  (N=33)
    {
        int n = NROWS * XPROJ_N;
        xproj_kernel<<<(n + TPB - 1) / TPB, TPB, 0, stream>>>(xcBf, W_xp, ssmF);
    }

    // 5) softplus / deltaB / C extraction
    {
        int n = NROWS * D_STATE;
        gate_prep_kernel<<<(n + TPB - 1) / TPB, TPB, 0, stream>>>(ssmF, dBBf, CmBf);
    }

    // 6) h_mean via WMMA (per batch: M=2048, N=16, K=2048)
    hmean_wmma_kernel<<<dim3(D_INNER / 64, BATCH), 128, 0, stream>>>(xcBf, dBBf, hmF);

    // 7) y recombination + SiLU(z) gating
    {
        int n = BATCH * SEQ * D_INNER;
        y_gate_kernel<<<(n + TPB - 1) / TPB, TPB, 0, stream>>>(hmF, CmBf, xcBf, Dv, xzBf, yBf);
    }

    // 8) out = y @ W_out  (M=8192, N=1024, K=2048), f32 out
    gemm_bf16_wmma<float>
        <<<dim3(D_MODEL / 128, NROWS / 128), 256, 0, stream>>>(
            yBf, WoutT, out, NROWS, D_MODEL, D_INNER);
}
